// SlotAttention_9139690405981
// MI455X (gfx1250) — matrix operand order repaired and checked by reading, not compiled, not run on hardware
//
#include <hip/hip_runtime.h>

typedef _Float16 half_t;
typedef __attribute__((ext_vector_type(16))) _Float16 v16h;
typedef __attribute__((ext_vector_type(8)))  _Float16 v8h;
typedef __attribute__((ext_vector_type(8)))  float    v8f;

#define BB   64
#define DIM  256
#define RES  32
#define NS   8
#define NTOK 1024
#define HID  256
#define ITERS 6            // 5 + 1 implicit-diff step (forward identical)
#define SCALE_ATT 0.0625f  // 256^-0.5
#define EPS_ATT 1e-8f
#define EPS_LN  1e-5f

// ---------------- epilogue modes for the unified WMMA GEMM ----------------
#define EP_F16      0   // f16 out = acc + bias
#define EP_F16_RELU 1   // f16 out = relu(acc + bias)
#define EP_F32_RELU 2   // f32 out = relu(acc + bias)
#define EP_F32      3   // f32 out = acc + bias
#define EP_VT       4   // f16 transposed store: out[b][n][t]  (for V^T)
#define EP_DOTS     5   // f32 out[b][slot][n] = acc*scale, slot<8 only
#define EP_UPD      6   // f16 compact out[(b*8+slot)][n], slot<8 only
#define EP_RES      7   // f32 out = acc + bias + res

// C[M,N] = A[M,K] @ W[N,K]^T (+bias, epilogue). 256 threads = 8 waves (2 wm x 4 wn).
// Each wave computes MT m-tiles x 2 n-tiles. K-loop is manually unrolled x2 with
// explicit ping/pong fragment registers (no rotation copies): each stage computes
// from one set while the other set's global_load_b128s are in flight, so the
// compiler can use partial s_wait_loadcnt values. Requires K % 64 == 0.
// If wBatchStride != 0, W is batched with 16 A-rows per batch (requires MT==1).
template <int MT>
__global__ __launch_bounds__(256) void gemm_wmma(
    const half_t* __restrict__ A, const half_t* __restrict__ W,
    long long wBatchStride, const float* __restrict__ bias,
    int K, int N, int mode, float scale,
    void* __restrict__ outv, const float* __restrict__ res)
{
  const int lane = threadIdx.x & 31;
  const int wave = threadIdx.x >> 5;
  const int wm = wave >> 2, wn = wave & 3;
  const int m0 = blockIdx.x * (32 * MT) + wm * (16 * MT);
  const int n0 = blockIdx.y * 128 + wn * 32;

  const half_t* Wb = W;
  if (wBatchStride) Wb += (size_t)(m0 >> 4) * (size_t)wBatchStride;

  const int half_ = lane >> 4;
  const int nA    = n0 + (lane & 15);

  // A operand, this code's K order (not AMD's table order): lane half selects contiguous K 16-chunk {h*16..h*16+15}
  const half_t* arow[MT];
#pragma unroll
  for (int mt = 0; mt < MT; ++mt)
    arow[mt] = A + (size_t)(m0 + mt * 16 + (lane & 15)) * K + half_ * 16;
  // B operand, same K order as A (not AMD's table order): lane half selects contiguous K 16-chunk {h*16..h*16+15}
  const half_t* w0 = Wb + (size_t)nA        * K + half_ * 16;
  const half_t* w1 = Wb + (size_t)(nA + 16) * K + half_ * 16;

  __builtin_prefetch(w0, 0, 3);             // global_prefetch_b8 (hoisted, once)
  __builtin_prefetch(arow[0], 0, 3);

  v8f acc[MT][2];
#pragma unroll
  for (int mt = 0; mt < MT; ++mt) { acc[mt][0] = (v8f){}; acc[mt][1] = (v8f){}; }

  v8h aloP[MT], ahiP[MT];   // ping
  v8h aloQ[MT], ahiQ[MT];   // pong
  v16h b0P, b1P, b0Q, b1Q;

  auto loadA = [&](v8h (&lo)[MT], v8h (&hi)[MT], int kk) {
#pragma unroll
    for (int mt = 0; mt < MT; ++mt) {
      lo[mt] = *(const v8h*)(arow[mt] + kk);
      hi[mt] = *(const v8h*)(arow[mt] + kk + 8);
    }
  };
  auto mmac = [&](const v8h (&lo)[MT], const v8h (&hi)[MT], v16h b0, v16h b1) {
#pragma unroll
    for (int mt = 0; mt < MT; ++mt) {
      v16h af;
#pragma unroll
      for (int i = 0; i < 8; ++i) { af[i] = lo[mt][i]; af[i + 8] = hi[mt][i]; }
      acc[mt][0] = __builtin_amdgcn_wmma_f32_16x16x32_f16(false, af, false, b0, (short)0, acc[mt][0], false, false);
      acc[mt][1] = __builtin_amdgcn_wmma_f32_16x16x32_f16(false, af, false, b1, (short)0, acc[mt][1], false, false);
    }
  };

  loadA(aloP, ahiP, 0);
  b0P = *(const v16h*)w0;
  b1P = *(const v16h*)w1;

  for (int k0 = 0; k0 < K; k0 += 64) {
    // stage A: prefetch pong (k0+32 always < K since K % 64 == 0), compute ping
    loadA(aloQ, ahiQ, k0 + 32);
    b0Q = *(const v16h*)(w0 + k0 + 32);
    b1Q = *(const v16h*)(w1 + k0 + 32);
    mmac(aloP, ahiP, b0P, b1P);
    // stage B: prefetch ping for next iteration, compute pong
    if (k0 + 64 < K) {
      loadA(aloP, ahiP, k0 + 64);
      b0P = *(const v16h*)(w0 + k0 + 64);
      b1P = *(const v16h*)(w1 + k0 + 64);
    }
    mmac(aloQ, ahiQ, b0Q, b1Q);
  }

  // C/D layout: lane = n + 16*(m>=8); vgpr e -> M = e + 8*(lane>=16)
  const int laneN = lane & 15;
  const int mAdd  = (lane >> 4) * 8;
#pragma unroll
  for (int mt = 0; mt < MT; ++mt) {
#pragma unroll
    for (int t = 0; t < 2; ++t) {
      const int n = n0 + t * 16 + laneN;
      const float bv = bias ? bias[n] : 0.f;
#pragma unroll
      for (int e = 0; e < 8; ++e) {
        const int m = m0 + mt * 16 + mAdd + e;
        float v = acc[mt][t][e] + bv;
        const size_t idx = (size_t)m * N + n;
        switch (mode) {
          case EP_F16:      ((half_t*)outv)[idx] = (half_t)v; break;
          case EP_F16_RELU: ((half_t*)outv)[idx] = (half_t)fmaxf(v, 0.f); break;
          case EP_F32_RELU: ((float*)outv)[idx]  = fmaxf(v, 0.f); break;
          case EP_F32:      ((float*)outv)[idx]  = v; break;
          case EP_RES:      ((float*)outv)[idx]  = v + res[idx]; break;
          case EP_VT: {
            const int b = m >> 10, tt = m & 1023;
            ((half_t*)outv)[(size_t)b * (size_t)N * 1024 + (size_t)n * 1024 + tt] = (half_t)v;
          } break;
          case EP_DOTS: {
            const int b = m >> 4, sl = m & 15;
            if (sl < 8) ((float*)outv)[(size_t)b * 8 * N + (size_t)sl * N + n] = v * scale;
          } break;
          case EP_UPD: {
            const int b = m >> 4, sl = m & 15;
            if (sl < 8) ((half_t*)outv)[((size_t)(b * 8 + sl)) * N + n] = (half_t)v;
          } break;
        }
      }
    }
  }
}

// transpose [B,D,T]->[B,T,D] via LDS tile, add soft position embed, feed ln2 stats
__global__ __launch_bounds__(256) void enc_stage0(
    const float* __restrict__ inp, const float* __restrict__ Wp,
    const float* __restrict__ bp, float* __restrict__ f0, float* __restrict__ sums)
{
  __shared__ float tile[32][33];
  const int tid = threadIdx.x;
  const int b = blockIdx.z, t0 = blockIdx.x * 32, d0 = blockIdx.y * 32;
  const int tl = tid & 31, th = tid >> 5;
#pragma unroll
  for (int i = 0; i < 4; ++i) {
    const int dl = th + 8 * i;
    tile[dl][tl] = inp[(size_t)b * DIM * NTOK + (size_t)(d0 + dl) * NTOK + (t0 + tl)];
  }
  __syncthreads();
  float s1 = 0.f, s2 = 0.f;
#pragma unroll
  for (int i = 0; i < 4; ++i) {
    const int tl2 = th + 8 * i, dl2 = tl;
    const int t = t0 + tl2, d = d0 + dl2;
    const int y = t >> 5, x = t & 31;
    const float gy = y * (1.f / 31.f), gx = x * (1.f / 31.f);
    const float pos = gy * Wp[d * 4 + 0] + gx * Wp[d * 4 + 1]
                    + (1.f - gy) * Wp[d * 4 + 2] + (1.f - gx) * Wp[d * 4 + 3] + bp[d];
    const float v = tile[dl2][tl2] + pos;
    f0[(size_t)b * NTOK * DIM + (size_t)t * DIM + d] = v;
    s1 += v; s2 += v * v;
  }
  __syncthreads();
  float* red = &tile[0][0];
  red[tid] = s1; red[256 + tid] = s2;
  __syncthreads();
  for (int s = 128; s > 0; s >>= 1) {
    if (tid < s) { red[tid] += red[tid + s]; red[256 + tid] += red[256 + tid + s]; }
    __syncthreads();
  }
  if (tid == 0) { atomicAdd(&sums[b], red[0]); atomicAdd(&sums[BB + b], red[256]); }
}

__global__ void apply_ln2(const float* __restrict__ f0, const float* __restrict__ sums,
                          const float* __restrict__ g, const float* __restrict__ bta,
                          half_t* __restrict__ out)
{
  const size_t i = (size_t)blockIdx.x * 256 + threadIdx.x;
  const int b = (int)(i >> 18);          // NTOK*DIM = 2^18
  const int td = (int)(i & 262143);
  const float mean = sums[b] * (1.f / 262144.f);
  const float var  = sums[BB + b] * (1.f / 262144.f) - mean * mean;
  const float rstd = rsqrtf(var + EPS_LN);
  out[i] = (half_t)((f0[i] - mean) * rstd * g[td] + bta[td]);
}

// per-row LN over 256, f32 in -> f16 out
__global__ __launch_bounds__(256) void ln_rows(
    const float* __restrict__ in, const float* __restrict__ g,
    const float* __restrict__ b, half_t* __restrict__ out)
{
  __shared__ float red[256];
  const int tid = threadIdx.x;
  const size_t row = blockIdx.x;
  const float x = in[row * 256 + tid];
  red[tid] = x; __syncthreads();
  for (int s = 128; s; s >>= 1) { if (tid < s) red[tid] += red[tid + s]; __syncthreads(); }
  const float m = red[0] * (1.f / 256.f); __syncthreads();
  const float dx = x - m;
  red[tid] = dx * dx; __syncthreads();
  for (int s = 128; s; s >>= 1) { if (tid < s) red[tid] += red[tid + s]; __syncthreads(); }
  const float rstd = rsqrtf(red[0] * (1.f / 256.f) + EPS_LN);
  out[row * 256 + tid] = (half_t)(dx * rstd * g[tid] + b[tid]);
}

// LN of slots -> Sh padded [B,16,256] (rows 8..15 zero) + raw slots f16 Hh [B*8,256]
__global__ __launch_bounds__(256) void ln_slots_k(
    const float* __restrict__ slots, const float* __restrict__ g,
    const float* __restrict__ b, half_t* __restrict__ Sh, half_t* __restrict__ Hh)
{
  const int tid = threadIdx.x;
  const int blk = blockIdx.x, bb = blk >> 4, sl = blk & 15;
  if (sl >= 8) { Sh[(size_t)blk * 256 + tid] = (half_t)0.f; return; }
  __shared__ float red[256];
  const size_t row = (size_t)bb * 8 + sl;
  const float x = slots[row * 256 + tid];
  red[tid] = x; __syncthreads();
  for (int s = 128; s; s >>= 1) { if (tid < s) red[tid] += red[tid + s]; __syncthreads(); }
  const float m = red[0] * (1.f / 256.f); __syncthreads();
  const float dx = x - m;
  red[tid] = dx * dx; __syncthreads();
  for (int s = 128; s; s >>= 1) { if (tid < s) red[tid] += red[tid + s]; __syncthreads(); }
  const float rstd = rsqrtf(red[0] * (1.f / 256.f) + EPS_LN);
  Sh[(size_t)blk * 256 + tid] = (half_t)(dx * rstd * g[tid] + b[tid]);
  Hh[row * 256 + tid] = (half_t)x;
}

// softmax over the 8 slots per (b, token), +EPS
__global__ void softmax_slots(const float* __restrict__ dots, float* __restrict__ attn)
{
  const int i = blockIdx.x * 256 + threadIdx.x;       // B*NTOK = 65536
  const int b = i >> 10, j = i & 1023;
  const float* dp = dots + (size_t)b * 8192 + j;
  float v[8], mx = -1e30f;
#pragma unroll
  for (int s = 0; s < 8; ++s) { v[s] = dp[s * 1024]; mx = fmaxf(mx, v[s]); }
  float sum = 0.f;
#pragma unroll
  for (int s = 0; s < 8; ++s) { v[s] = expf(v[s] - mx); sum += v[s]; }
  const float inv = 1.f / sum;
  float* ap = attn + (size_t)b * 8192 + j;
#pragma unroll
  for (int s = 0; s < 8; ++s) ap[s * 1024] = v[s] * inv + EPS_ATT;
}

// normalize each slot row over its 1024 tokens -> f16 A-matrix [B,16,1024] (pad rows zeroed)
__global__ __launch_bounds__(256) void attn_norm(const float* __restrict__ attn,
                                                 half_t* __restrict__ Ah)
{
  const int blk = blockIdx.x, b = blk >> 4, sl = blk & 15;
  const int tid = threadIdx.x;
  half_t* out = Ah + (size_t)blk * 1024;
  if (sl >= 8) {
#pragma unroll
    for (int i = 0; i < 4; ++i) out[tid + 256 * i] = (half_t)0.f;
    return;
  }
  const float* ap = attn + ((size_t)b * 8 + sl) * 1024;
  __shared__ float red[256];
  float lv[4]; float s = 0.f;
#pragma unroll
  for (int i = 0; i < 4; ++i) { lv[i] = ap[tid + 256 * i]; s += lv[i]; }
  red[tid] = s; __syncthreads();
  for (int st = 128; st; st >>= 1) { if (tid < st) red[tid] += red[tid + st]; __syncthreads(); }
  const float inv = 1.f / red[0];
#pragma unroll
  for (int i = 0; i < 4; ++i) out[tid + 256 * i] = (half_t)(lv[i] * inv);
}

__global__ void gru_fuse(const float* __restrict__ GI, const float* __restrict__ GH,
                         float* __restrict__ slots)
{
  const int i = blockIdx.x * 256 + threadIdx.x;       // 512*256
  const int r = i >> 8, d = i & 255;
  const float* gi = GI + (size_t)r * 768;
  const float* gh = GH + (size_t)r * 768;
  const float ir = gi[d],       hr = gh[d];
  const float iz = gi[256 + d], hz = gh[256 + d];
  const float in_ = gi[512 + d], hn = gh[512 + d];
  const float rg = 1.f / (1.f + expf(-(ir + hr)));
  const float z  = 1.f / (1.f + expf(-(iz + hz)));
  const float n  = tanhf(in_ + rg * hn);
  const float h  = slots[i];
  slots[i] = (1.f - z) * n + z * h;
}

__global__ void init_slots_k(const float* __restrict__ mu, const float* __restrict__ sigma,
                             const float* __restrict__ noise, float* __restrict__ slots)
{
  const int i = blockIdx.x * 256 + threadIdx.x;       // B*NS*DIM
  const int d = i & 255;
  slots[i] = mu[d] + sigma[d] * noise[i];
}

__global__ void cvt_f16(const float* __restrict__ in, half_t* __restrict__ out, int n)
{
  const int i = blockIdx.x * 256 + threadIdx.x;
  if (i < n) out[i] = (half_t)in[i];
}

__global__ void zerof(float* __restrict__ p, int n)
{
  const int i = blockIdx.x * 256 + threadIdx.x;
  if (i < n) p[i] = 0.f;
}

extern "C" void kernel_launch(void* const* d_in, const int* in_sizes, int n_in,
                              void* d_out, int out_size, void* d_ws, size_t ws_size,
                              hipStream_t stream) {
  const float* inputs = (const float*)d_in[0];
  const float* noise  = (const float*)d_in[1];
  const float* W_pos  = (const float*)d_in[2];
  const float* b_pos  = (const float*)d_in[3];
  const float* g_enc  = (const float*)d_in[4];
  const float* b_enc  = (const float*)d_in[5];
  const float* W_m1   = (const float*)d_in[6];
  const float* b_m1   = (const float*)d_in[7];
  const float* W_m2   = (const float*)d_in[8];
  const float* b_m2   = (const float*)d_in[9];
  const float* g_in   = (const float*)d_in[10];
  const float* b_in   = (const float*)d_in[11];
  const float* Wq     = (const float*)d_in[12];
  const float* bq     = (const float*)d_in[13];
  const float* Wk     = (const float*)d_in[14];
  const float* bk     = (const float*)d_in[15];
  const float* Wv     = (const float*)d_in[16];
  const float* bv     = (const float*)d_in[17];
  const float* W_ih   = (const float*)d_in[18];
  const float* W_hh   = (const float*)d_in[19];
  const float* b_ih   = (const float*)d_in[20];
  const float* b_hh   = (const float*)d_in[21];
  const float* g_s    = (const float*)d_in[22];
  const float* b_s    = (const float*)d_in[23];
  const float* g_ff   = (const float*)d_in[24];
  const float* b_ff   = (const float*)d_in[25];
  const float* W_f1   = (const float*)d_in[26];
  const float* b_f1   = (const float*)d_in[27];
  const float* W_f2   = (const float*)d_in[28];
  const float* b_f2   = (const float*)d_in[29];
  const float* mu     = (const float*)d_in[30];
  const float* sigma  = (const float*)d_in[31];

  char* wsb = (char*)d_ws;
  size_t off = 0;
  auto alloc = [&](size_t bytes) -> void* {
    void* p = wsb + off;
    off = (off + bytes + 255) & ~(size_t)255;
    return p;
  };
  float*  sums = (float*)alloc(2 * BB * sizeof(float));
  float*  bufA = (float*)alloc((size_t)BB * NTOK * DIM * 4);   // f0, then X2
  half_t* bufB = (half_t*)alloc((size_t)BB * NTOK * DIM * 2);  // X0h, then Fh
  half_t* bufC = (half_t*)alloc((size_t)BB * NTOK * DIM * 2);  // X1h, then Kh
  half_t* Vth  = (half_t*)alloc((size_t)BB * DIM * NTOK * 2);  // V^T per batch [256,1024]
  half_t* Wm1h = (half_t*)alloc(DIM * DIM * 2);
  half_t* Wm2h = (half_t*)alloc(DIM * DIM * 2);
  half_t* Wqh  = (half_t*)alloc(DIM * DIM * 2);
  half_t* Wkh  = (half_t*)alloc(DIM * DIM * 2);
  half_t* Wvh  = (half_t*)alloc(DIM * DIM * 2);
  half_t* Wihh = (half_t*)alloc(3 * DIM * DIM * 2);
  half_t* Whhh = (half_t*)alloc(3 * DIM * DIM * 2);
  half_t* Wf1h = (half_t*)alloc(HID * DIM * 2);
  half_t* Wf2h = (half_t*)alloc(DIM * HID * 2);
  float*  slots = (float*)alloc((size_t)BB * NS * DIM * 4);
  half_t* Sh   = (half_t*)alloc((size_t)BB * 16 * DIM * 2);
  half_t* Hh   = (half_t*)alloc((size_t)BB * NS * DIM * 2);
  half_t* Qh   = (half_t*)alloc((size_t)BB * 16 * DIM * 2);
  float*  dots = (float*)alloc((size_t)BB * NS * NTOK * 4);
  float*  attn = (float*)alloc((size_t)BB * NS * NTOK * 4);
  half_t* Ah   = (half_t*)alloc((size_t)BB * 16 * NTOK * 2);
  half_t* Xh   = (half_t*)alloc((size_t)BB * NS * DIM * 2);
  float*  GI   = (float*)alloc((size_t)BB * NS * 3 * DIM * 4);
  float*  GH   = (float*)alloc((size_t)BB * NS * 3 * DIM * 4);
  half_t* Th   = (half_t*)alloc((size_t)BB * NS * DIM * 2);
  half_t* Hf   = (half_t*)alloc((size_t)BB * NS * HID * 2);
  (void)ws_size; (void)n_in; (void)in_sizes; (void)out_size;

  // MT=2 (wave computes 32x32) for all shared-weight GEMMs; M must be %64.
  auto gemm2 = [&](const half_t* A, const half_t* W, const float* bias,
                   int M, int K, int N, int mode, void* out, const float* res) {
    dim3 g(M / 64, N / 128);
    gemm_wmma<2><<<g, 256, 0, stream>>>(A, W, 0, bias, K, N, mode, 1.f, out, res);
  };
  // MT=1 (wave stays inside one 16-row batch) for per-batch-B attention GEMMs.
  auto gemm1b = [&](const half_t* A, const half_t* W, long long wstride,
                    int M, int K, int N, int mode, float scale, void* out) {
    dim3 g(M / 32, N / 128);
    gemm_wmma<1><<<g, 256, 0, stream>>>(A, W, wstride, nullptr, K, N, mode, scale, out, nullptr);
  };
  auto cvt = [&](const float* s, half_t* d, int n) {
    cvt_f16<<<(n + 255) / 256, 256, 0, stream>>>(s, d, n);
  };

  // ---- weight conversion (f32 -> f16) ----
  cvt(W_m1, Wm1h, DIM * DIM);   cvt(W_m2, Wm2h, DIM * DIM);
  cvt(Wq,   Wqh,  DIM * DIM);   cvt(Wk,   Wkh,  DIM * DIM);
  cvt(Wv,   Wvh,  DIM * DIM);
  cvt(W_ih, Wihh, 3 * DIM * DIM); cvt(W_hh, Whhh, 3 * DIM * DIM);
  cvt(W_f1, Wf1h, HID * DIM);   cvt(W_f2, Wf2h, DIM * HID);

  // ---- encoder ----
  zerof<<<1, 256, 0, stream>>>(sums, 2 * BB);
  enc_stage0<<<dim3(32, 8, BB), 256, 0, stream>>>(inputs, W_pos, b_pos, bufA, sums);
  apply_ln2<<<65536, 256, 0, stream>>>(bufA, sums, g_enc, b_enc, bufB);       // X0h
  gemm2(bufB, Wm1h, b_m1, BB * NTOK, DIM, DIM, EP_F16_RELU, bufC, nullptr);   // X1h
  gemm2(bufC, Wm2h, b_m2, BB * NTOK, DIM, DIM, EP_F32_RELU, bufA, nullptr);   // X2
  ln_rows<<<BB * NTOK, 256, 0, stream>>>(bufA, g_in, b_in, bufB);             // Fh
  gemm2(bufB, Wkh, bk, BB * NTOK, DIM, DIM, EP_F16, bufC, nullptr);           // Kh
  gemm2(bufB, Wvh, bv, BB * NTOK, DIM, DIM, EP_VT, Vth, nullptr);             // V^T

  // ---- slot init + iterations ----
  init_slots_k<<<(BB * NS * DIM) / 256, 256, 0, stream>>>(mu, sigma, noise, slots);
  for (int it = 0; it < ITERS; ++it) {
    ln_slots_k<<<BB * 16, 256, 0, stream>>>(slots, g_s, b_s, Sh, Hh);
    gemm2(Sh, Wqh, bq, BB * 16, DIM, DIM, EP_F16, Qh, nullptr);
    gemm1b(Qh, bufC, (long long)NTOK * DIM, BB * 16, DIM, NTOK, EP_DOTS, SCALE_ATT, dots);
    softmax_slots<<<(BB * NTOK) / 256, 256, 0, stream>>>(dots, attn);
    attn_norm<<<BB * 16, 256, 0, stream>>>(attn, Ah);
    gemm1b(Ah, Vth, (long long)DIM * NTOK, BB * 16, NTOK, DIM, EP_UPD, 1.f, Xh);
    gemm2(Xh, Wihh, b_ih, BB * NS, DIM, 3 * DIM, EP_F32, GI, nullptr);
    gemm2(Hh, Whhh, b_hh, BB * NS, DIM, 3 * DIM, EP_F32, GH, nullptr);
    gru_fuse<<<(BB * NS * DIM) / 256, 256, 0, stream>>>(GI, GH, slots);
    ln_rows<<<BB * NS, 256, 0, stream>>>(slots, g_ff, b_ff, Th);
    gemm2(Th, Wf1h, b_f1, BB * NS, DIM, HID, EP_F16_RELU, Hf, nullptr);
    gemm2(Hf, Wf2h, b_f2, BB * NS, HID, DIM, EP_RES, slots, slots);
  }

  hipMemcpyAsync(d_out, slots, (size_t)BB * NS * DIM * sizeof(float),
                 hipMemcpyDeviceToDevice, stream);
}
